// MultiHeadSA_57698590654770
// MI455X (gfx1250) — matrix operand order repaired and checked by reading, not compiled, not run on hardware
//
#include <hip/hip_runtime.h>

// Problem constants (from reference)
#define NH    16
#define DM    1024
#define HS    64
#define BB    2
#define TT    2048
#define NQKV  3072
#define MROWS (BB * TT)   // 4096 token rows

typedef __bf16 bf16;
typedef __attribute__((ext_vector_type(16))) __bf16 v16bf;
typedef __attribute__((ext_vector_type(8)))  float  v8f;
typedef __attribute__((ext_vector_type(4)))  unsigned int v4u;
typedef __attribute__((ext_vector_type(8)))  unsigned int v8u;

union FragU { v16bf v; uint4 u[2]; };

// ---- LDS byte offset of a __shared__ object (strip flat aperture) ----
__device__ __forceinline__ unsigned lds_addr_of(const void* p) {
    return (unsigned)(unsigned long long)(__attribute__((address_space(3))) const void*)p;
}

// ---- Tensor Data Mover: 2-D tile load, bf16 elements, LDS row padding ----
// D# per CDNA5 ISA.
// pad_int: 3 -> pad after 16 DWORDs (32 bf16 row), 4 -> after 32 DWORDs (64 bf16 row)
// pad_amt: 3 -> 4 DWORDs (8 bf16) of padding  => strides 40 / 72 bf16
__device__ __forceinline__ void tdm_load_2d(unsigned lds_off, const void* gptr,
                                            unsigned td0, unsigned td1, unsigned stride0,
                                            unsigned tile0, unsigned tile1,
                                            unsigned pad_int, unsigned pad_amt) {
    unsigned long long ga = (unsigned long long)gptr;
    v4u g0;
    g0[0] = 1u;                                    // count=1, user mode, no gather
    g0[1] = lds_off;                               // lds_addr
    g0[2] = (unsigned)ga;                          // global_addr[31:0]
    g0[3] = ((unsigned)(ga >> 32) & 0x01ffffffu)   // global_addr[56:32]
          | 0x80000000u;                           // type = 2 ("image")
    v8u g1;
    g1[0] = (1u << 16)                             // data_size = 1 (2 bytes)
          | (1u << 20)                             // pad_enable
          | (pad_int << 22) | (pad_amt << 25);
    g1[1] = (td0 & 0xffffu) << 16;                 // tensor_dim0[15:0]
    g1[2] = (td0 >> 16) | ((td1 & 0xffffu) << 16); // tensor_dim0[31:16] | tensor_dim1[15:0]
    g1[3] = (td1 >> 16) | (tile0 << 16);           // tensor_dim1[31:16] | tile_dim0
    g1[4] = tile1;                                 // tile_dim1 (tile_dim2 = 0)
    g1[5] = stride0;                               // tensor_dim0_stride[31:0]
    g1[6] = 0u;                                    // stride0[47:32]=0 | dim1_stride[15:0]=0
    g1[7] = 0u;
    asm volatile("tensor_load_to_lds %0, %1" :: "s"(g0), "s"(g1) : "memory");
}

// ---- WMMA fragment loaders (layouts per CDNA5 ISA 7.12.2, wave32) ----
__device__ __forceinline__ v16bf load_a_frag(const bf16* base, int stride, int lane) {
    int hl = lane >> 4, lr = lane & 15;
    const bf16* p = base + lr * stride + hl * 8;
    FragU f;
    f.u[0] = *(const uint4*)(p);        // k = hl*8 .. hl*8+7
    f.u[1] = *(const uint4*)(p + 16);   // k = 16+hl*8 ..
    return f.v;
}
__device__ __forceinline__ v16bf load_b_frag(const bf16* base, int stride, int lane) {
    int hl = lane >> 4, lr = lane & 15;
    const bf16* p = base + lr * stride + hl * 8;
    FragU f;
    f.u[0] = *(const uint4*)(p);
    f.u[1] = *(const uint4*)(p + 16);
    return f.v;
}
// C/D 16x16 f32: c[v] = C[v + 8*half][lane&15]

// ------------------------------------------------------------------
// Kernel 0: one-time f32 -> bf16 conversion; weights stored transposed
// ------------------------------------------------------------------
__global__ void convert_kernel(const float* __restrict__ X,
                               const float* __restrict__ Wqkv,
                               const float* __restrict__ Wo,
                               bf16* __restrict__ Xb,
                               bf16* __restrict__ WqkvT,
                               bf16* __restrict__ WoT) {
    int tid = blockIdx.x * blockDim.x + threadIdx.x;
    if (tid < MROWS * DM) Xb[tid] = (bf16)X[tid];
    if (tid < NQKV * DM) {              // WqkvT[n][k] = Wqkv[k][n]
        int n = tid >> 10, k = tid & 1023;
        WqkvT[tid] = (bf16)Wqkv[(size_t)k * NQKV + n];
    }
    if (tid < DM * DM) {                // WoT[n][k] = Wo[k][n]
        int n = tid >> 10, k = tid & 1023;
        WoT[tid] = (bf16)Wo[(size_t)k * DM + n];
    }
}

// ------------------------------------------------------------------
// Kernel 1: fused QKV GEMM via TDM double-buffered pipeline.
// ------------------------------------------------------------------
#define ASTR 40
#define BSTR 40

__global__ __launch_bounds__(128) void qkv_gemm_kernel(
        const bf16* __restrict__ Xb, const bf16* __restrict__ WT,
        const float* __restrict__ bqkv,
        bf16* __restrict__ Qb, bf16* __restrict__ Kb, bf16* __restrict__ Vt) {
    __shared__ __align__(16) bf16 Asl[2][64 * ASTR];
    __shared__ __align__(16) bf16 Bsl[2][64 * BSTR];
    int lane = threadIdx.x & 31, wave = threadIdx.x >> 5;
    int hl = lane >> 4, lr = lane & 15;
    int mbase = blockIdx.y * 64, nbase = blockIdx.x * 64;
    int wm = wave >> 1, wn = wave & 1;
    v8f acc[2][2] = {};

    const bf16* Ag = Xb + (size_t)mbase * DM;
    const bf16* Bg = WT + (size_t)nbase * DM;

    if (wave == 0) {
        tdm_load_2d(lds_addr_of(&Asl[0][0]), Ag, DM, MROWS, DM, 32, 64, 3, 3);
        tdm_load_2d(lds_addr_of(&Bsl[0][0]), Bg, DM, NQKV,  DM, 32, 64, 3, 3);
    }

    const int NK = DM / 32;
    for (int it = 0; it < NK; ++it) {
        int cur = it & 1;
        if (wave == 0) __builtin_amdgcn_s_wait_tensorcnt(0);
        __syncthreads();                       // buf[cur] ready; buf[cur^1] free
        if (wave == 0 && it + 1 < NK) {        // DMA next tiles under the WMMAs
            tdm_load_2d(lds_addr_of(&Asl[cur ^ 1][0]), Ag + (it + 1) * 32,
                        DM, MROWS, DM, 32, 64, 3, 3);
            tdm_load_2d(lds_addr_of(&Bsl[cur ^ 1][0]), Bg + (it + 1) * 32,
                        DM, NQKV,  DM, 32, 64, 3, 3);
        }
        v16bf a0 = load_a_frag(&Asl[cur][(32 * wm +  0) * ASTR], ASTR, lane);
        v16bf a1 = load_a_frag(&Asl[cur][(32 * wm + 16) * ASTR], ASTR, lane);
        v16bf b0 = load_b_frag(&Bsl[cur][(32 * wn +  0) * BSTR], BSTR, lane);
        v16bf b1 = load_b_frag(&Bsl[cur][(32 * wn + 16) * BSTR], BSTR, lane);
        acc[0][0] = __builtin_amdgcn_wmma_f32_16x16x32_bf16(false, a0, false, b0, (short)0, acc[0][0], false, false);
        acc[0][1] = __builtin_amdgcn_wmma_f32_16x16x32_bf16(false, a0, false, b1, (short)0, acc[0][1], false, false);
        acc[1][0] = __builtin_amdgcn_wmma_f32_16x16x32_bf16(false, a1, false, b0, (short)0, acc[1][0], false, false);
        acc[1][1] = __builtin_amdgcn_wmma_f32_16x16x32_bf16(false, a1, false, b1, (short)0, acc[1][1], false, false);
    }

    #pragma unroll
    for (int i = 0; i < 2; ++i) {
        #pragma unroll
        for (int j = 0; j < 2; ++j) {
            int col = nbase + 32 * wn + 16 * j + lr;
            float bias = bqkv[col];
            int h = col / 192, s = col % 192;
            #pragma unroll
            for (int v = 0; v < 8; ++v) {
                int row = mbase + 32 * wm + 16 * i + v + 8 * hl;
                int bi = row >> 11, t = row & (TT - 1);
                bf16 bv = (bf16)(acc[i][j][v] + bias);
                size_t hb = (size_t)(bi * NH + h);
                if (s < 64)       Qb[(hb * TT + t) * HS + s] = bv;
                else if (s < 128) Kb[(hb * TT + t) * HS + (s - 64)] = bv;
                else              Vt[(hb * HS + (s - 128)) * TT + t] = bv;
            }
        }
    }
}

// ------------------------------------------------------------------
// Kernel 2: flash attention, TDM double-buffered K/V streaming.
// ------------------------------------------------------------------
#define QSTR 72

__global__ __launch_bounds__(128) void attn_kernel(
        const bf16* __restrict__ Qb, const bf16* __restrict__ Kb,
        const bf16* __restrict__ Vt, const float* __restrict__ mask,
        bf16* __restrict__ Ob) {
    __shared__ __align__(16) bf16 Qs[64 * QSTR];
    __shared__ __align__(16) bf16 Ks[2][64 * QSTR];
    __shared__ __align__(16) bf16 Vs[2][64 * QSTR];
    __shared__ __align__(16) bf16 Ps[64 * QSTR];
    int lane = threadIdx.x & 31, wave = threadIdx.x >> 5;
    int hl = lane >> 4, lr = lane & 15;
    int bh = blockIdx.y, bi = bh / NH, h = bh % NH;
    const bf16* Qp = Qb + (size_t)bh * TT * HS;
    const bf16* Kp = Kb + (size_t)bh * TT * HS;
    const bf16* Vp = Vt + (size_t)bh * HS * TT;
    int qbase = blockIdx.x * 64;

    if (wave == 0) {  // Q block + first K/V blocks via TDM
        tdm_load_2d(lds_addr_of(&Qs[0]),       Qp + (size_t)qbase * HS, HS, TT, HS, 64, 64, 4, 3);
        tdm_load_2d(lds_addr_of(&Ks[0][0]),    Kp,                      HS, TT, HS, 64, 64, 4, 3);
        tdm_load_2d(lds_addr_of(&Vs[0][0]),    Vp,                      TT, HS, TT, 64, 64, 4, 3);
    }

    float mrow[8], lrow[8];
    v8f oacc[4] = {};
    #pragma unroll
    for (int v = 0; v < 8; ++v) { mrow[v] = -1e30f; lrow[v] = 0.0f; }

    const int NB = TT / 64;
    for (int it = 0; it < NB; ++it) {
        int cur = it & 1, kb = it * 64;
        if (wave == 0) __builtin_amdgcn_s_wait_tensorcnt(0);
        __syncthreads();
        if (wave == 0 && it + 1 < NB) {     // stream next K/V under compute
            tdm_load_2d(lds_addr_of(&Ks[cur ^ 1][0]), Kp + (size_t)(kb + 64) * HS,
                        HS, TT, HS, 64, 64, 4, 3);
            tdm_load_2d(lds_addr_of(&Vs[cur ^ 1][0]), Vp + (kb + 64),
                        TT, HS, TT, 64, 64, 4, 3);
        }

        // S = Q * K^T (64 kv cols, K-depth 64 in 2 steps)
        v8f sacc[4] = {};
        #pragma unroll
        for (int ks = 0; ks < 2; ++ks) {
            v16bf a = load_a_frag(&Qs[(16 * wave) * QSTR + 32 * ks], QSTR, lane);
            #pragma unroll
            for (int j = 0; j < 4; ++j) {
                v16bf b = load_b_frag(&Ks[cur][(16 * j) * QSTR + 32 * ks], QSTR, lane);
                sacc[j] = __builtin_amdgcn_wmma_f32_16x16x32_bf16(false, a, false, b, (short)0, sacc[j], false, false);
            }
        }

        // online softmax per query row
        float pv[4][8];
        #pragma unroll
        for (int v = 0; v < 8; ++v) {
            int qrow = qbase + 16 * wave + v + 8 * hl;
            const float* mp = mask + ((size_t)bi * TT + qrow) * TT + kb + lr;
            float rmax = -1e30f;
            #pragma unroll
            for (int j = 0; j < 4; ++j) {
                float x = sacc[j][v] * 0.125f + mp[16 * j];
                pv[j][v] = x;
                rmax = fmaxf(rmax, x);
            }
            #pragma unroll
            for (int d = 1; d < 16; d <<= 1)
                rmax = fmaxf(rmax, __shfl_xor(rmax, d, 32));
            float mnew = fmaxf(mrow[v], rmax);
            float fsc  = __expf(mrow[v] - mnew);
            float rsum = 0.0f;
            #pragma unroll
            for (int j = 0; j < 4; ++j) {
                float e = __expf(pv[j][v] - mnew);
                pv[j][v] = e;
                rsum += e;
            }
            #pragma unroll
            for (int d = 1; d < 16; d <<= 1)
                rsum += __shfl_xor(rsum, d, 32);
            lrow[v] = lrow[v] * fsc + rsum;
            mrow[v] = mnew;
            #pragma unroll
            for (int jo = 0; jo < 4; ++jo) oacc[jo][v] *= fsc;
            #pragma unroll
            for (int j = 0; j < 4; ++j)   // wave-private Ps rows; DS in-order per wave
                Ps[(16 * wave + v + 8 * hl) * QSTR + 16 * j + lr] = (bf16)pv[j][v];
        }

        // O += P * V  (Vs[n=d][k=kv])
        #pragma unroll
        for (int ks = 0; ks < 2; ++ks) {
            v16bf a = load_a_frag(&Ps[(16 * wave) * QSTR + 32 * ks], QSTR, lane);
            #pragma unroll
            for (int jo = 0; jo < 4; ++jo) {
                v16bf b = load_b_frag(&Vs[cur][(16 * jo) * QSTR + 32 * ks], QSTR, lane);
                oacc[jo] = __builtin_amdgcn_wmma_f32_16x16x32_bf16(false, a, false, b, (short)0, oacc[jo], false, false);
            }
        }
    }

    #pragma unroll
    for (int v = 0; v < 8; ++v) {
        int qrow = qbase + 16 * wave + v + 8 * hl;
        float inv = 1.0f / lrow[v];
        size_t ro = ((size_t)(bi * TT + qrow)) * DM + h * HS;
        #pragma unroll
        for (int jo = 0; jo < 4; ++jo)
            Ob[ro + 16 * jo + lr] = (bf16)(oacc[jo][v] * inv);
    }
}

// ------------------------------------------------------------------
// Kernel 3: output projection via TDM double-buffered pipeline.
// ------------------------------------------------------------------
__global__ __launch_bounds__(128) void proj_gemm_kernel(
        const bf16* __restrict__ Ab, const bf16* __restrict__ WT,
        const float* __restrict__ bias, float* __restrict__ out) {
    __shared__ __align__(16) bf16 Asl[2][64 * ASTR];
    __shared__ __align__(16) bf16 Bsl[2][64 * BSTR];
    int lane = threadIdx.x & 31, wave = threadIdx.x >> 5;
    int hl = lane >> 4, lr = lane & 15;
    int mbase = blockIdx.y * 64, nbase = blockIdx.x * 64;
    int wm = wave >> 1, wn = wave & 1;
    v8f acc[2][2] = {};

    const bf16* Ag = Ab + (size_t)mbase * DM;
    const bf16* Bg = WT + (size_t)nbase * DM;

    if (wave == 0) {
        tdm_load_2d(lds_addr_of(&Asl[0][0]), Ag, DM, MROWS, DM, 32, 64, 3, 3);
        tdm_load_2d(lds_addr_of(&Bsl[0][0]), Bg, DM, DM,    DM, 32, 64, 3, 3);
    }

    const int NK = DM / 32;
    for (int it = 0; it < NK; ++it) {
        int cur = it & 1;
        if (wave == 0) __builtin_amdgcn_s_wait_tensorcnt(0);
        __syncthreads();
        if (wave == 0 && it + 1 < NK) {
            tdm_load_2d(lds_addr_of(&Asl[cur ^ 1][0]), Ag + (it + 1) * 32,
                        DM, MROWS, DM, 32, 64, 3, 3);
            tdm_load_2d(lds_addr_of(&Bsl[cur ^ 1][0]), Bg + (it + 1) * 32,
                        DM, DM,    DM, 32, 64, 3, 3);
        }
        v16bf a0 = load_a_frag(&Asl[cur][(32 * wm +  0) * ASTR], ASTR, lane);
        v16bf a1 = load_a_frag(&Asl[cur][(32 * wm + 16) * ASTR], ASTR, lane);
        v16bf b0 = load_b_frag(&Bsl[cur][(32 * wn +  0) * BSTR], BSTR, lane);
        v16bf b1 = load_b_frag(&Bsl[cur][(32 * wn + 16) * BSTR], BSTR, lane);
        acc[0][0] = __builtin_amdgcn_wmma_f32_16x16x32_bf16(false, a0, false, b0, (short)0, acc[0][0], false, false);
        acc[0][1] = __builtin_amdgcn_wmma_f32_16x16x32_bf16(false, a0, false, b1, (short)0, acc[0][1], false, false);
        acc[1][0] = __builtin_amdgcn_wmma_f32_16x16x32_bf16(false, a1, false, b0, (short)0, acc[1][0], false, false);
        acc[1][1] = __builtin_amdgcn_wmma_f32_16x16x32_bf16(false, a1, false, b1, (short)0, acc[1][1], false, false);
    }

    #pragma unroll
    for (int i = 0; i < 2; ++i) {
        #pragma unroll
        for (int j = 0; j < 2; ++j) {
            int col = nbase + 32 * wn + 16 * j + lr;
            float bv = bias[col];
            #pragma unroll
            for (int v = 0; v < 8; ++v) {
                int row = mbase + 32 * wm + 16 * i + v + 8 * hl;
                out[(size_t)row * DM + col] = acc[i][j][v] + bv;
            }
        }
    }
}

// ------------------------------------------------------------------
extern "C" void kernel_launch(void* const* d_in, const int* in_sizes, int n_in,
                              void* d_out, int out_size, void* d_ws, size_t ws_size,
                              hipStream_t stream) {
    (void)in_sizes; (void)n_in; (void)out_size; (void)ws_size;
    const float* X    = (const float*)d_in[0];
    const float* mask = (const float*)d_in[1];
    const float* Wqkv = (const float*)d_in[2];
    const float* bqkv = (const float*)d_in[3];
    const float* Wo   = (const float*)d_in[4];
    const float* bo   = (const float*)d_in[5];
    float* out = (float*)d_out;

    char* ws = (char*)d_ws;
    bf16* Xb    = (bf16*)(ws);                 //  8 MB  [4096,1024]
    bf16* WqkvT = (bf16*)(ws + ( 8ull << 20)); //  6 MB  [3072,1024]
    bf16* WoT   = (bf16*)(ws + (14ull << 20)); //  2 MB  [1024,1024]
    bf16* Qb    = (bf16*)(ws + (16ull << 20)); //  8 MB  [B,H,T,hs]
    bf16* Kb    = (bf16*)(ws + (24ull << 20)); //  8 MB  [B,H,T,hs]
    bf16* Vt    = (bf16*)(ws + (32ull << 20)); //  8 MB  [B,H,hs,T]
    bf16* Ob    = (bf16*)(ws + (40ull << 20)); //  8 MB  [4096,1024]

    convert_kernel<<<(MROWS * DM) / 256, 256, 0, stream>>>(X, Wqkv, Wo, Xb, WqkvT, WoT);

    qkv_gemm_kernel<<<dim3(NQKV / 64, MROWS / 64), 128, 0, stream>>>(
        Xb, WqkvT, bqkv, Qb, Kb, Vt);

    attn_kernel<<<dim3(TT / 64, BB * NH), 128, 0, stream>>>(Qb, Kb, Vt, mask, Ob);

    proj_gemm_kernel<<<dim3(DM / 64, MROWS / 64), 128, 0, stream>>>(Ob, WoT, bo, out);
}
